// Head_2388001816842
// MI455X (gfx1250) — compile-verified, matrix operand order checked by reading, not run on hardware
//
#include <hip/hip_runtime.h>
#include <hip/hip_bf16.h>

typedef __attribute__((ext_vector_type(16))) _Float16 v16h;
typedef __attribute__((ext_vector_type(8)))  _Float16 h8;
typedef __attribute__((ext_vector_type(8)))  float    v8f;

#define BB 4
#define TT 4096
#define CC 1024
#define HH 64
#define KSPLIT 4                                            // waves per query tile

static __device__ __forceinline__ v8f wmma_f16(v16h a, v16h b, v8f c) {
    return __builtin_amdgcn_wmma_f32_16x16x32_f16(false, a, false, b, (short)0, c, false, false);
}

// Load a 16-element f16 fragment: 8 f16 at p+c0, 8 f16 at p+c1 (two b128 loads)
static __device__ __forceinline__ v16h ld16(const _Float16* p, int c0, int c1) {
    h8 lo = *(const h8*)(p + c0);
    h8 hi = *(const h8*)(p + c1);
    v16h r;
#pragma unroll
    for (int i = 0; i < 8; ++i) { r[i] = lo[i]; r[i + 8] = hi[i]; }
    return r;
}

// ---------------- Kernel 1: convert + transpose weights: Wt[m][h][c] = (f16)W_m[c][h]
__global__ void k_prep_w(const float* __restrict__ Wk, const float* __restrict__ Wq,
                         const float* __restrict__ Wv, _Float16* __restrict__ Wt) {
    int idx = blockIdx.x * blockDim.x + threadIdx.x;
    if (idx >= 3 * HH * CC) return;
    int m   = idx / (HH * CC);
    int rem = idx % (HH * CC);
    int h = rem / CC, c = rem % CC;
    const float* W = (m == 0) ? Wq : (m == 1) ? Wk : Wv;   // 0=Q, 1=K, 2=V
    Wt[idx] = (_Float16)W[c * HH + h];
}

// ---------------- Kernel 2: projections. One wave per (matrix, 16-token tile).
// Q,K stored row-major [B*T][H] f16; V stored transposed Vt[B][H][T] f16.
__global__ void __launch_bounds__(256) k_proj(const float* __restrict__ x,
                                              const _Float16* __restrict__ Wt,
                                              _Float16* __restrict__ Qh,
                                              _Float16* __restrict__ Kh,
                                              _Float16* __restrict__ Vt) {
    const int NTILE = (BB * TT) / 16;                       // 1024 token tiles
    int wid  = (blockIdx.x * blockDim.x + threadIdx.x) >> 5;
    int lane = threadIdx.x & 31;
    if (wid >= 3 * NTILE) return;
    int mat  = wid / NTILE;
    int tt   = wid % NTILE;
    int tok0 = tt * 16;
    int row   = lane & 15;
    int halfo = (lane >> 4) << 3;                           // 0 or 8

    const _Float16* Wm = Wt + (size_t)mat * HH * CC;
    v8f acc[4] = {v8f{}, v8f{}, v8f{}, v8f{}};

    for (int k0 = 0; k0 < CC; k0 += 32) {
        // A fragment: 16 tokens x 32 channels, converted f32 -> f16 in-register
        const float* xr = x + (size_t)(tok0 + row) * CC + k0;
        v16h a;
#pragma unroll
        for (int i = 0; i < 8; ++i) {
            a[i]     = (_Float16)xr[halfo + i];
            a[i + 8] = (_Float16)xr[16 + halfo + i];
        }
#pragma unroll
        for (int f = 0; f < 4; ++f) {
            const _Float16* wr = Wm + (size_t)(f * 16 + row) * CC + k0;
            v16h b = ld16(wr, halfo, 16 + halfo);
            acc[f] = wmma_f16(a, b, acc[f]);
        }
    }

#pragma unroll
    for (int f = 0; f < 4; ++f) {
#pragma unroll
        for (int r = 0; r < 8; ++r) {
            int tok = tok0 + halfo + r;                     // D: m = r + 8*(lane>>4)
            int h   = f * 16 + row;                         //    n = lane%16
            _Float16 val = (_Float16)acc[f][r];
            if (mat == 0)      Qh[(size_t)tok * HH + h] = val;
            else if (mat == 1) Kh[(size_t)tok * HH + h] = val;
            else {
                int bb = tok >> 12, tl = tok & (TT - 1);
                Vt[((size_t)bb * HH + h) * TT + tl] = val;
            }
        }
    }
}

// ---------------- Kernel 3: causal flash attention with key-split.
// One 4-wave workgroup per (batch, 16-query tile); wave s takes key blocks
// j = s*32 + k*128. Partials (m, l, Oacc) combined exactly through LDS.
__global__ void __launch_bounds__(32 * KSPLIT) k_attn(const _Float16* __restrict__ Qh,
                                                      const _Float16* __restrict__ Kh,
                                                      const _Float16* __restrict__ Vt,
                                                      float* __restrict__ out) {
    const int QT = TT / 16;                                 // 256 query tiles per batch
    int b  = blockIdx.x / QT;
    int qt = blockIdx.x % QT;
    int q0 = qt * 16;
    int wv   = threadIdx.x >> 5;                            // key-split wave id 0..3
    int lane = threadIdx.x & 31;
    int row   = lane & 15;                                  // query column (B-operand / D n)
    int halfo = (lane >> 4) << 3;                           // 0 or 8

    __shared__ float lds_o[KSPLIT][4][8][32];               // [wave][hfrag][r][lane] 16 KB
    __shared__ float lds_m[KSPLIT][32];
    __shared__ float lds_l[KSPLIT][32];

    const _Float16* Qb = Qh + (size_t)b * TT * HH;
    const _Float16* Kb = Kh + (size_t)b * TT * HH;
    const _Float16* Vb = Vt + (size_t)b * HH * TT;

    // Q fragments (B operand), kept in registers for the whole key loop
    const _Float16* qr = Qb + (size_t)(q0 + row) * HH;
    v16h qf0 = ld16(qr,      halfo, 16 + halfo);            // h 0..31
    v16h qf1 = ld16(qr + 32, halfo, 16 + halfo);            // h 32..63

    float mrun = -1e30f, lrun = 0.0f;
    v8f o[4] = {v8f{}, v8f{}, v8f{}, v8f{}};                // O^T partials, h tiles of 16

    const int kend = q0 + 16;                               // exclusive causal key bound
    for (int j = wv * 32; j < kend; j += 32 * KSPLIT) {
        // S^T = K * Q^T for two 16-key subtiles (keys j..j+15 and j+16..j+31)
        v8f s[2];
#pragma unroll
        for (int t = 0; t < 2; ++t) {
            const _Float16* kr = Kb + (size_t)(j + t * 16 + row) * HH;
            v16h kf0 = ld16(kr,      halfo, 16 + halfo);
            v16h kf1 = ld16(kr + 32, halfo, 16 + halfo);
            v8f sa = {};
            sa = wmma_f16(kf0, qf0, sa);
            sa = wmma_f16(kf1, qf1, sa);
            s[t] = sa;
        }

        // scale by C^-0.5 = 1/32, causal mask, per-query running max
        float tmax = -1e30f;
#pragma unroll
        for (int t = 0; t < 2; ++t) {
#pragma unroll
            for (int r = 0; r < 8; ++r) {
                int kg  = j + t * 16 + halfo + r;           // global key index
                float v = s[t][r] * 0.03125f;
                v = (kg <= q0 + row) ? v : -1e30f;
                s[t][r] = v;
                tmax = fmaxf(tmax, v);
            }
        }
        tmax = fmaxf(tmax, __shfl_xor(tmax, 16, 32));       // combine key halves
        float mnew  = fmaxf(mrun, tmax);
        float scale = __expf(mrun - mnew);

        // exp + row sum; pack P directly into the f16 B-operand layout for PV
        float rowsum = 0.0f;
        v16h pf;
#pragma unroll
        for (int t = 0; t < 2; ++t) {
#pragma unroll
            for (int r = 0; r < 8; ++r) {
                float p = __expf(s[t][r] - mnew);
                rowsum += p;
                pf[t * 8 + r] = (_Float16)p;
            }
        }
        rowsum += __shfl_xor(rowsum, 16, 32);
        lrun = lrun * scale + rowsum;
        mrun = mnew;

        // rescale accumulators, then O^T += V^T * P^T (4 h-tiles of 16)
#pragma unroll
        for (int f = 0; f < 4; ++f) {
#pragma unroll
            for (int r = 0; r < 8; ++r) o[f][r] *= scale;
            const _Float16* vr = Vb + (size_t)(f * 16 + row) * TT + j;
            v16h vf = ld16(vr, halfo, 16 + halfo);          // 32 keys of V^T row h
            o[f] = wmma_f16(vf, pf, o[f]);
        }
    }

    // publish partials
#pragma unroll
    for (int f = 0; f < 4; ++f)
#pragma unroll
        for (int r = 0; r < 8; ++r) lds_o[wv][f][r][lane] = o[f][r];
    lds_m[wv][lane] = mrun;
    lds_l[wv][lane] = lrun;
    __syncthreads();

    // exact combine: wave wv finalizes h-fragment f == wv
    float mtot = -1e30f;
#pragma unroll
    for (int s2 = 0; s2 < KSPLIT; ++s2) mtot = fmaxf(mtot, lds_m[s2][lane]);
    float w[KSPLIT];
    float ltot = 0.0f;
#pragma unroll
    for (int s2 = 0; s2 < KSPLIT; ++s2) {
        w[s2] = __expf(lds_m[s2][lane] - mtot);
        ltot += lds_l[s2][lane] * w[s2];
    }
    float inv = 1.0f / ltot;

    float* ob = out + ((size_t)b * TT + q0) * HH;           // out[b][q][h]
#pragma unroll
    for (int r = 0; r < 8; ++r) {
        float acc = 0.0f;
#pragma unroll
        for (int s2 = 0; s2 < KSPLIT; ++s2) acc += lds_o[s2][wv][r][lane] * w[s2];
        ob[(size_t)row * HH + wv * 16 + halfo + r] = acc * inv;
    }
}

extern "C" void kernel_launch(void* const* d_in, const int* in_sizes, int n_in,
                              void* d_out, int out_size, void* d_ws, size_t ws_size,
                              hipStream_t stream) {
    const float* x  = (const float*)d_in[0];
    const float* Wk = (const float*)d_in[1];
    const float* Wq = (const float*)d_in[2];
    const float* Wv = (const float*)d_in[3];
    float* out = (float*)d_out;

    // workspace carve-up (f16): Wt[3][H][C] | Qh[B*T][H] | Kh[B*T][H] | Vt[B][H][T]
    _Float16* Wt = (_Float16*)d_ws;
    _Float16* Qh = Wt + (size_t)3 * HH * CC;
    _Float16* Kh = Qh + (size_t)BB * TT * HH;
    _Float16* Vt = Kh + (size_t)BB * TT * HH;

    {   // weight transpose/convert: 3*64*1024 elements
        int n = 3 * HH * CC;
        k_prep_w<<<(n + 255) / 256, 256, 0, stream>>>(Wk, Wq, Wv, Wt);
    }
    {   // projections: 3 matrices * 1024 token tiles = 3072 waves, 8 waves/block
        int waves = 3 * (BB * TT / 16);
        k_proj<<<(waves + 7) / 8, 256, 0, stream>>>(x, Wt, Qh, Kh, Vt);
    }
    {   // attention: one 4-wave block per (batch, qtile) = 1024 blocks, 4096 waves
        k_attn<<<BB * (TT / 16), 32 * KSPLIT, 0, stream>>>(Qh, Kh, Vt, out);
    }
}
